// NMF_8220567404862
// MI455X (gfx1250) — matrix operand order repaired and checked by reading, not compiled, not run on hardware
//
#include <hip/hip_runtime.h>

// ---------------------------------------------------------------------------
// Batched NMF multiplicative updates on MI455X (gfx1250, wave32, WMMA).
// All GEMMs use v_wmma_f32_16x16x32_bf16 with a split-bf16 (hi+lo) scheme:
//   a*b ~= ah*bh + ah*bl + al*bh   (3 WMMAs, f32 accumulate, ~fp32 accuracy)
// Splits are precomputed once (x) / carried across iterations (D, C) so the
// hot K-loops are pure copy-staging + WMMA. LDS tiles are double-buffered
// (ping-pong) so staging of chunk c+1 overlaps WMMA of chunk c with a single
// workgroup barrier per chunk.
// ---------------------------------------------------------------------------

typedef __bf16 bf16;
typedef __attribute__((ext_vector_type(16))) __bf16 v16bf;
typedef __attribute__((ext_vector_type(8)))  __bf16 v8bf;
typedef __attribute__((ext_vector_type(4)))  __bf16 v4bf;
typedef __attribute__((ext_vector_type(8)))  float  v8f;

#define KC   32          // K elements per LDS tile (one WMMA K-step)
#define LDK  40          // padded LDS row stride in bf16 elems (16B-aligned rows)
#define TILE (64 * LDK)  // one 64x32 tile in LDS
#define EPSF 2e-8f

// ---- WMMA helpers ---------------------------------------------------------

__device__ __forceinline__ v8f wmma_bf16(v16bf a, v16bf b, v8f c) {
  return __builtin_amdgcn_wmma_f32_16x16x32_bf16(false, a, false, b, (short)0, c,
                                                 false, false);
}

__device__ __forceinline__ v8f mma3(v16bf ah, v16bf al, v16bf bh, v16bf bl, v8f c) {
  c = wmma_bf16(ah, bl, c);   // small cross terms first
  c = wmma_bf16(al, bh, c);
  c = wmma_bf16(ah, bh, c);   // dominant term
  return c;
}

__device__ __forceinline__ void splitf(float f, bf16& h, bf16& l) {
  h = (bf16)f;
  l = (bf16)(f - (float)h);
}

// ---- f32-source staging (fallback path: converts while staging) -----------
// 64(row) x 32(K) tile; dst[i][k], row stride LDK. 512 threads, one float4 each.

__device__ __forceinline__ void stage_direct_f32(bf16* dh, bf16* dl,
                                                 const float* src, int sld,
                                                 int r0, int k0, bool do_relu) {
  const int idx = threadIdx.x;
  const int i = idx >> 3, k = (idx & 7) << 2;
  float4 v = *(const float4*)(src + (size_t)(r0 + i) * sld + (k0 + k));
  if (do_relu) {
    v.x = fmaxf(v.x, 0.0f); v.y = fmaxf(v.y, 0.0f);
    v.z = fmaxf(v.z, 0.0f); v.w = fmaxf(v.w, 0.0f);
  }
  const float a[4] = {v.x, v.y, v.z, v.w};
  bf16* ph = dh + i * LDK + k;
  bf16* pl = dl + i * LDK + k;
#pragma unroll
  for (int t = 0; t < 4; ++t) { bf16 h, l; splitf(a[t], h, l); ph[t] = h; pl[t] = l; }
}

__device__ __forceinline__ void stage_transp_f32(bf16* dh, bf16* dl,
                                                 const float* src, int sld,
                                                 int r0, int k0, bool do_relu) {
  const int idx = threadIdx.x;
  const int k = idx >> 4, i = (idx & 15) << 2;
  float4 v = *(const float4*)(src + (size_t)(k0 + k) * sld + (r0 + i));
  if (do_relu) {
    v.x = fmaxf(v.x, 0.0f); v.y = fmaxf(v.y, 0.0f);
    v.z = fmaxf(v.z, 0.0f); v.w = fmaxf(v.w, 0.0f);
  }
  const float a[4] = {v.x, v.y, v.z, v.w};
#pragma unroll
  for (int t = 0; t < 4; ++t) {
    bf16 h, l; splitf(a[t], h, l);
    dh[(i + t) * LDK + k] = h;
    dl[(i + t) * LDK + k] = l;
  }
}

// ---- bf16-source staging (fast path: pure copies) -------------------------
// Threads 0..255 stage the hi array, 256..511 the lo array (8 elems each).

__device__ __forceinline__ void stage_direct_bf(bf16* dh, bf16* dl,
                                                const bf16* sh, const bf16* sl,
                                                int sld, int r0, int k0) {
  const int idx = threadIdx.x;
  const int sel = idx >> 8, t = idx & 255;
  const int i = t >> 2, k = (t & 3) << 3;           // 64 rows x 4 oct-K
  const bf16* s = sel ? sl : sh;
  bf16*       d = sel ? dl : dh;
  v8bf v = *(const v8bf*)(s + (size_t)(r0 + i) * sld + (k0 + k));
  *(v8bf*)(d + i * LDK + k) = v;
}

__device__ __forceinline__ void stage_transp_bf(bf16* dh, bf16* dl,
                                                const bf16* sh, const bf16* sl,
                                                int sld, int r0, int k0) {
  const int idx = threadIdx.x;
  const int sel = idx >> 8, t = idx & 255;
  const int k = t >> 3, i = (t & 7) << 3;           // 32 K x 8 oct-rows
  const bf16* s = sel ? sl : sh;
  bf16*       d = sel ? dl : dh;
  v8bf v = *(const v8bf*)(s + (size_t)(k0 + k) * sld + (r0 + i));
#pragma unroll
  for (int e = 0; e < 8; ++e) d[(i + e) * LDK + k] = v[e];
}

// ---- Fragment loads (ISA 16-bit A 16x32 / B 32x16 lane layouts) -----------

__device__ __forceinline__ v16bf load_a_frag(const bf16* base, int m0, int lane) {
  const int M  = m0 + (lane & 15);
  const int kb = (lane & 16) >> 1;                 // 0 or 8
  const bf16* p = base + M * LDK;
  v8bf lo = *(const v8bf*)(p + kb);                // K = kb..kb+7
  v8bf hi = *(const v8bf*)(p + 16 + kb);           // K = 16+kb..16+kb+7
  return __builtin_shufflevector(lo, hi, 0, 1, 2, 3, 4, 5, 6, 7,
                                 8, 9, 10, 11, 12, 13, 14, 15);
}

__device__ __forceinline__ v16bf load_b_frag(const bf16* base, int n0, int lane) {
  const int N  = n0 + (lane & 15);
  const int kb = (lane & 16) >> 1;                 // 0 or 8
  const bf16* p = base + N * LDK;
  v8bf lo = *(const v8bf*)(p + kb);
  v8bf hi = *(const v8bf*)(p + 16 + kb);
  return __builtin_shufflevector(lo, hi, 0, 1, 2, 3, 4, 5, 6, 7,
                                 8, 9, 10, 11, 12, 13, 14, 15);
}

// ---------------------------------------------------------------------------
// Prep: split f32 -> (hi, lo) bf16, optional relu. One float4 per thread.
// ---------------------------------------------------------------------------
__global__ __launch_bounds__(256) void split_kernel(bf16* __restrict__ h,
                                                    bf16* __restrict__ l,
                                                    const float* __restrict__ src,
                                                    int do_relu, long n4) {
  const long idx = (long)blockIdx.x * 256 + threadIdx.x;
  if (idx >= n4) return;
  float4 v = ((const float4*)src)[idx];
  if (do_relu) {
    v.x = fmaxf(v.x, 0.0f); v.y = fmaxf(v.y, 0.0f);
    v.z = fmaxf(v.z, 0.0f); v.w = fmaxf(v.w, 0.0f);
  }
  const float a[4] = {v.x, v.y, v.z, v.w};
  v4bf hv, lv;
#pragma unroll
  for (int t = 0; t < 4; ++t) { bf16 hh, ll; splitf(a[t], hh, ll); hv[t] = hh; lv[t] = ll; }
  ((v4bf*)h)[idx] = hv;
  ((v4bf*)l)[idx] = lv;
}

// Ping-pong GEMM phase driver: stage(buf,k0) must stage chunk k0 into buffer
// slot buf; comp(buf,acc) must accumulate one chunk. One barrier per chunk.
template <typename STAGE, typename COMP>
__device__ __forceinline__ void pipeline(int nchunks, STAGE&& stage, COMP&& comp,
                                         v8f& acc) {
  stage(0, 0);
  __syncthreads();
  for (int c = 0; c < nchunks; ++c) {
    const int cur = c & 1;
    if (c + 1 < nchunks) stage(cur ^ 1, (c + 1) * KC);
    acc = comp(cur, acc);
    __syncthreads();
  }
}

// ---------------------------------------------------------------------------
// Kernel 1: 64x64 Gram matrix per batch.
//   TR=1: out = P^T P (P is [Ktot,64]; lds[i][k]=P[k][i])  -> D^T D
//   TR=0: out = P P^T (P is [64,Ktot]; lds[i][k]=P[i][k])  -> C C^T
// A and B fragments share one LDS tile. grid = B, block = 512 (16 waves)
// ---------------------------------------------------------------------------
template <bool TR, bool SPLIT>
__global__ __launch_bounds__(512) void gram_kernel(
    float* __restrict__ out, const float* __restrict__ P,
    const bf16* __restrict__ Psh, const bf16* __restrict__ Psl,
    long bstride, int sld, int Ktot) {
  __shared__ __align__(16) bf16 Ph[2][TILE];
  __shared__ __align__(16) bf16 Pl[2][TILE];
  const size_t bo = (size_t)blockIdx.x * bstride;
  const int lane = threadIdx.x & 31, wave = threadIdx.x >> 5;
  const int m0 = (wave >> 2) * 16, n0 = (wave & 3) * 16;
  v8f acc = {};
  auto stage = [&](int buf, int k0) {
    if constexpr (SPLIT) {
      if constexpr (TR) stage_transp_bf(Ph[buf], Pl[buf], Psh + bo, Psl + bo, sld, 0, k0);
      else              stage_direct_bf(Ph[buf], Pl[buf], Psh + bo, Psl + bo, sld, 0, k0);
    } else {
      if constexpr (TR) stage_transp_f32(Ph[buf], Pl[buf], P + bo, sld, 0, k0, false);
      else              stage_direct_f32(Ph[buf], Pl[buf], P + bo, sld, 0, k0, false);
    }
  };
  auto comp = [&](int buf, v8f a) {
    return mma3(load_a_frag(Ph[buf], m0, lane), load_a_frag(Pl[buf], m0, lane),
                load_b_frag(Ph[buf], n0, lane), load_b_frag(Pl[buf], n0, lane), a);
  };
  pipeline(Ktot / KC, stage, comp, acc);

  float* ob = out + (size_t)blockIdx.x * 4096;
  const int N = n0 + (lane & 15);
#pragma unroll
  for (int j = 0; j < 8; ++j) {
    const int M = m0 + j + ((lane & 16) >> 1);
    ob[M * 64 + N] = acc[j];
  }
}

// ---------------------------------------------------------------------------
// Kernel 2: C_new = C .* (D^T x) ./ (G C + eps), per batch, 64 x 2048.
// grid = (2048/64, B). Also emits split-bf16 copy of C_new for the next pass.
// ---------------------------------------------------------------------------
template <bool SPLIT>
__global__ __launch_bounds__(512) void updateC_kernel(
    float* __restrict__ Cnew, bf16* __restrict__ Cnh, bf16* __restrict__ Cnl,
    const float* __restrict__ x, const bf16* __restrict__ xh, const bf16* __restrict__ xl,
    const float* __restrict__ Dold, const bf16* __restrict__ Dh, const bf16* __restrict__ Dl, long sD,
    const float* __restrict__ Cold, const bf16* __restrict__ Ch, const bf16* __restrict__ Cl, long sC,
    const float* __restrict__ G) {
  __shared__ __align__(16) bf16 Ah[2][TILE], Al[2][TILE];
  __shared__ __align__(16) bf16 Bh[2][TILE], Bl[2][TILE];
  const int b  = blockIdx.y;
  const int nb = blockIdx.x * 64;
  const size_t xo = (size_t)b * 512 * 2048;
  const size_t doff = (size_t)b * sD, coff = (size_t)b * sC;
  const float* Cb = Cold + coff;
  const float* Gb = G + (size_t)b * 4096;
  const int lane = threadIdx.x & 31, wave = threadIdx.x >> 5;
  const int m0 = (wave >> 2) * 16, n0 = (wave & 3) * 16;

  v8f accN = {}, accD = {};
  auto comp = [&](int buf, v8f a) {
    return mma3(load_a_frag(Ah[buf], m0, lane), load_a_frag(Al[buf], m0, lane),
                load_b_frag(Bh[buf], n0, lane), load_b_frag(Bl[buf], n0, lane), a);
  };
  // numerator (D^T x): A[m=r][k=d] = D[d][r]; B[n][k] = relu(x)[d=k][nb+n]
  auto stageN = [&](int buf, int k0) {
    if constexpr (SPLIT) {
      stage_transp_bf(Ah[buf], Al[buf], Dh + doff, Dl + doff, 64, 0, k0);
      stage_transp_bf(Bh[buf], Bl[buf], xh + xo, xl + xo, 2048, nb, k0);
    } else {
      stage_transp_f32(Ah[buf], Al[buf], Dold + doff, 64, 0, k0, false);
      stage_transp_f32(Bh[buf], Bl[buf], x + xo, 2048, nb, k0, true);
    }
  };
  pipeline(512 / KC, stageN, comp, accN);
  // denominator (G C): A[m][k] = G[m][k]; B[n][k] = C[k][nb+n]
  auto stageD = [&](int buf, int k0) {
    stage_direct_f32(Ah[buf], Al[buf], Gb, 64, 0, k0, false);
    if constexpr (SPLIT) stage_transp_bf(Bh[buf], Bl[buf], Ch + coff, Cl + coff, 2048, nb, k0);
    else                 stage_transp_f32(Bh[buf], Bl[buf], Cb, 2048, nb, k0, false);
  };
  pipeline(64 / KC, stageD, comp, accD);

  const size_t obase = (size_t)b * (64 * 2048);
  const int N = nb + n0 + (lane & 15);
#pragma unroll
  for (int j = 0; j < 8; ++j) {
    const int M = m0 + j + ((lane & 16) >> 1);
    const size_t off = (size_t)M * 2048 + N;
    const float val = Cb[off] * accN[j] / (accD[j] + EPSF);
    Cnew[obase + off] = val;
    if constexpr (SPLIT) {
      bf16 h, l; splitf(val, h, l);
      Cnh[obase + off] = h;
      Cnl[obase + off] = l;
    }
  }
}

// ---------------------------------------------------------------------------
// Kernel 3: D_new = D .* (x C^T) ./ (D H + eps), per batch, 512 x 64.
// grid = (512/64, B). Also emits split-bf16 copy of D_new.
// ---------------------------------------------------------------------------
template <bool SPLIT>
__global__ __launch_bounds__(512) void updateD_kernel(
    float* __restrict__ Dnew, bf16* __restrict__ Dnh, bf16* __restrict__ Dnl,
    const float* __restrict__ x, const bf16* __restrict__ xh, const bf16* __restrict__ xl,
    const float* __restrict__ Cold, const bf16* __restrict__ Ch, const bf16* __restrict__ Cl, long sC,
    const float* __restrict__ Dold, const bf16* __restrict__ Dh, const bf16* __restrict__ Dl, long sD,
    const float* __restrict__ H) {
  __shared__ __align__(16) bf16 Ah[2][TILE], Al[2][TILE];
  __shared__ __align__(16) bf16 Bh[2][TILE], Bl[2][TILE];
  const int b  = blockIdx.y;
  const int mb = blockIdx.x * 64;
  const size_t xo = (size_t)b * 512 * 2048;
  const size_t doff = (size_t)b * sD, coff = (size_t)b * sC;
  const float* Db = Dold + doff;
  const float* Hb = H + (size_t)b * 4096;
  const int lane = threadIdx.x & 31, wave = threadIdx.x >> 5;
  const int m0 = (wave >> 2) * 16, n0 = (wave & 3) * 16;

  v8f accN = {}, accD = {};
  auto comp = [&](int buf, v8f a) {
    return mma3(load_a_frag(Ah[buf], m0, lane), load_a_frag(Al[buf], m0, lane),
                load_b_frag(Bh[buf], n0, lane), load_b_frag(Bl[buf], n0, lane), a);
  };
  // numerator (x C^T): A[m][k] = relu(x)[mb+m][k]; B[j=r][k=n] = C[j][k]
  auto stageN = [&](int buf, int k0) {
    if constexpr (SPLIT) {
      stage_direct_bf(Ah[buf], Al[buf], xh + xo, xl + xo, 2048, mb, k0);
      stage_direct_bf(Bh[buf], Bl[buf], Ch + coff, Cl + coff, 2048, 0, k0);
    } else {
      stage_direct_f32(Ah[buf], Al[buf], x + xo, 2048, mb, k0, true);
      stage_direct_f32(Bh[buf], Bl[buf], Cold + coff, 2048, 0, k0, false);
    }
  };
  pipeline(2048 / KC, stageN, comp, accN);
  // denominator (D H): A[m][k] = D[mb+m][k]; B[j][k] = H[k][j]
  auto stageD = [&](int buf, int k0) {
    if constexpr (SPLIT) stage_direct_bf(Ah[buf], Al[buf], Dh + doff, Dl + doff, 64, mb, k0);
    else                 stage_direct_f32(Ah[buf], Al[buf], Db, 64, mb, k0, false);
    stage_transp_f32(Bh[buf], Bl[buf], Hb, 64, 0, k0, false);
  };
  pipeline(64 / KC, stageD, comp, accD);

  const size_t obase = (size_t)b * (512 * 64);
  const int N = n0 + (lane & 15);
#pragma unroll
  for (int j = 0; j < 8; ++j) {
    const int M = mb + m0 + j + ((lane & 16) >> 1);
    const size_t off = (size_t)M * 64 + N;
    const float val = Db[off] * accN[j] / (accD[j] + EPSF);
    Dnew[obase + off] = val;
    if constexpr (SPLIT) {
      bf16 h, l; splitf(val, h, l);
      Dnh[obase + off] = h;
      Dnl[obase + off] = l;
    }
  }
}

// ---------------------------------------------------------------------------
// Kernel 4: out = D @ C, per batch, 512 x 2048 (K = 64). grid = (8, 32, B).
// ---------------------------------------------------------------------------
template <bool SPLIT>
__global__ __launch_bounds__(512) void outDC_kernel(
    float* __restrict__ out,
    const float* __restrict__ Dfin, const bf16* __restrict__ Dh, const bf16* __restrict__ Dl, long sD,
    const float* __restrict__ Cfin, const bf16* __restrict__ Ch, const bf16* __restrict__ Cl, long sC) {
  __shared__ __align__(16) bf16 Ah[2][TILE], Al[2][TILE];
  __shared__ __align__(16) bf16 Bh[2][TILE], Bl[2][TILE];
  const int b  = blockIdx.z;
  const int mb = blockIdx.x * 64;
  const int nb = blockIdx.y * 64;
  const size_t doff = (size_t)b * sD, coff = (size_t)b * sC;
  const int lane = threadIdx.x & 31, wave = threadIdx.x >> 5;
  const int m0 = (wave >> 2) * 16, n0 = (wave & 3) * 16;

  v8f acc = {};
  auto stage = [&](int buf, int k0) {
    if constexpr (SPLIT) {
      stage_direct_bf(Ah[buf], Al[buf], Dh + doff, Dl + doff, 64, mb, k0);     // A=D rows
      stage_transp_bf(Bh[buf], Bl[buf], Ch + coff, Cl + coff, 2048, nb, k0);   // B=C^T
    } else {
      stage_direct_f32(Ah[buf], Al[buf], Dfin + doff, 64, mb, k0, false);
      stage_transp_f32(Bh[buf], Bl[buf], Cfin + coff, 2048, nb, k0, false);
    }
  };
  auto comp = [&](int buf, v8f a) {
    return mma3(load_a_frag(Ah[buf], m0, lane), load_a_frag(Al[buf], m0, lane),
                load_b_frag(Bh[buf], n0, lane), load_b_frag(Bl[buf], n0, lane), a);
  };
  pipeline(64 / KC, stage, comp, acc);

  float* ob = out + (size_t)b * 512 * 2048;
  const int N = nb + n0 + (lane & 15);
#pragma unroll
  for (int j = 0; j < 8; ++j) {
    const int M = mb + m0 + j + ((lane & 16) >> 1);
    ob[(size_t)M * 2048 + N] = acc[j];
  }
}

// ---------------------------------------------------------------------------
extern "C" void kernel_launch(void* const* d_in, const int* in_sizes, int n_in,
                              void* d_out, int out_size, void* d_ws, size_t ws_size,
                              hipStream_t stream) {
  (void)in_sizes; (void)n_in; (void)out_size;
  const float* x  = (const float*)d_in[0];   // [32,512,2048]
  const float* D0 = (const float*)d_in[1];   // [512,64]  (broadcast over batch)
  const float* C0 = (const float*)d_in[2];   // [64,2048] (broadcast over batch)
  float* out = (float*)d_out;
  float* ws  = (float*)d_ws;

  const size_t szD  = (size_t)32 * 512 * 64;
  const size_t szC  = (size_t)32 * 64 * 2048;
  const size_t szG  = (size_t)32 * 64 * 64;
  const size_t szX  = (size_t)32 * 512 * 2048;
  const size_t szD0 = (size_t)512 * 64;
  const size_t szC0 = (size_t)64 * 2048;

  // f32 region
  float* Dbuf[2] = { ws, ws + szD };
  float* Cbuf[2] = { ws + 2 * szD, ws + 2 * szD + szC };
  float* G = ws + 2 * szD + 2 * szC;
  float* H = G + szG;
  const size_t f32_elems = 2 * szD + 2 * szC + 2 * szG;

  // bf16 region (fast path only)
  bf16* bb = (bf16*)(ws + f32_elems);
  bf16* xh = bb;                bf16* xl = xh + szX;
  bf16* Dbh[2] = { xl + szX,        xl + szX + szD };
  bf16* Dbl[2] = { Dbh[1] + szD,    Dbh[1] + 2 * szD };
  bf16* Cbh[2] = { Dbl[1] + szD,    Dbl[1] + szD + szC };
  bf16* Cbl[2] = { Cbh[1] + szC,    Cbh[1] + 2 * szC };
  bf16* D0h = Cbl[1] + szC;     bf16* D0l = D0h + szD0;
  bf16* C0h = D0l + szD0;       bf16* C0l = C0h + szC0;
  const size_t bf16_elems = 2 * szX + 4 * szD + 4 * szC + 2 * (szD0 + szC0);
  const size_t need = f32_elems * 4 + bf16_elems * 2;   // ~220 MB
  const bool SPL = ws_size >= need;

  if (SPL) {
    split_kernel<<<(unsigned)(szX / 4 / 256), 256, 0, stream>>>(xh, xl, x, 1, (long)(szX / 4));
    split_kernel<<<(unsigned)(szD0 / 4 / 256), 256, 0, stream>>>(D0h, D0l, D0, 0, (long)(szD0 / 4));
    split_kernel<<<(unsigned)(szC0 / 4 / 256), 256, 0, stream>>>(C0h, C0l, C0, 0, (long)(szC0 / 4));
  }

  const float* Dcur = D0; const bf16 *Dhc = D0h, *Dlc = D0l; long sD = 0;
  const float* Ccur = C0; const bf16 *Chc = C0h, *Clc = C0l; long sC = 0;
  int cur = 0;
  for (int it = 0; it < 6; ++it) {
    if (SPL) {
      gram_kernel<true, true><<<32, 512, 0, stream>>>(G, Dcur, Dhc, Dlc, sD, 64, 512);
      gram_kernel<false, true><<<32, 512, 0, stream>>>(H, Ccur, Chc, Clc, sC, 2048, 2048);
      updateC_kernel<true><<<dim3(32, 32), 512, 0, stream>>>(
          Cbuf[cur], Cbh[cur], Cbl[cur], x, xh, xl,
          Dcur, Dhc, Dlc, sD, Ccur, Chc, Clc, sC, G);
      updateD_kernel<true><<<dim3(8, 32), 512, 0, stream>>>(
          Dbuf[cur], Dbh[cur], Dbl[cur], x, xh, xl,
          Ccur, Chc, Clc, sC, Dcur, Dhc, Dlc, sD, H);
    } else {
      gram_kernel<true, false><<<32, 512, 0, stream>>>(G, Dcur, nullptr, nullptr, sD, 64, 512);
      gram_kernel<false, false><<<32, 512, 0, stream>>>(H, Ccur, nullptr, nullptr, sC, 2048, 2048);
      updateC_kernel<false><<<dim3(32, 32), 512, 0, stream>>>(
          Cbuf[cur], nullptr, nullptr, x, nullptr, nullptr,
          Dcur, nullptr, nullptr, sD, Ccur, nullptr, nullptr, sC, G);
      updateD_kernel<false><<<dim3(8, 32), 512, 0, stream>>>(
          Dbuf[cur], nullptr, nullptr, x, nullptr, nullptr,
          Ccur, nullptr, nullptr, sC, Dcur, nullptr, nullptr, sD, H);
    }
    Dcur = Dbuf[cur]; Dhc = Dbh[cur]; Dlc = Dbl[cur]; sD = 512 * 64;
    Ccur = Cbuf[cur]; Chc = Cbh[cur]; Clc = Cbl[cur]; sC = 64 * 2048;
    cur ^= 1;
  }
  if (SPL) {
    outDC_kernel<true><<<dim3(8, 32, 32), 512, 0, stream>>>(
        out, Dcur, Dhc, Dlc, sD, Ccur, Chc, Clc, sC);
  } else {
    outDC_kernel<false><<<dim3(8, 32, 32), 512, 0, stream>>>(
        out, Dcur, nullptr, nullptr, sD, Ccur, nullptr, nullptr, sC);
  }
}
